// MultiHeadAttention_60739427500252
// MI455X (gfx1250) — compile-verified, matrix operand order checked by reading, not run on hardware
//
#include <hip/hip_runtime.h>

// Problem constants (match reference)
#define BB 4
#define SS 2048
#define DD 1024
#define HH 16
#define HD 64

typedef __attribute__((ext_vector_type(16))) __bf16 v16bf;
typedef __attribute__((ext_vector_type(8)))  float  v8f;

union Frag {
  v16bf v;
  uint4 q[2];
};

__device__ __forceinline__ unsigned short f2bf(float f) {
  unsigned int u = __float_as_uint(f);
  u += 0x7fffu + ((u >> 16) & 1u);   // round-to-nearest-even
  return (unsigned short)(u >> 16);
}

__device__ __forceinline__ v8f wmma_bf16(const Frag& a, const Frag& b, v8f c) {
  // D = A(16x32 bf16) x B(32x16 bf16) + C(16x16 f32)
  return __builtin_amdgcn_wmma_f32_16x16x32_bf16(
      /*neg_a=*/false, a.v, /*neg_b=*/false, b.v,
      /*c_mod=*/(short)0, c, /*reuse_a=*/false, /*reuse_b=*/false);
}

// ---- 16-lane xor reductions: DPP ROW_XMASK (VALU-only) with shuffle fallback
#if __has_builtin(__builtin_amdgcn_update_dpp)
#define DPP_XOR(x_, mask_)                                                   \
  __builtin_bit_cast(float, __builtin_amdgcn_update_dpp(                     \
      0, __builtin_bit_cast(int, (x_)), 0x160 + (mask_), 0xf, 0xf, true))
__device__ __forceinline__ float rowmax16(float x) {
  x = fmaxf(x, DPP_XOR(x, 1));
  x = fmaxf(x, DPP_XOR(x, 2));
  x = fmaxf(x, DPP_XOR(x, 4));
  x = fmaxf(x, DPP_XOR(x, 8));
  return x;
}
__device__ __forceinline__ float rowsum16(float x) {
  x += DPP_XOR(x, 1);
  x += DPP_XOR(x, 2);
  x += DPP_XOR(x, 4);
  x += DPP_XOR(x, 8);
  return x;
}
#else
__device__ __forceinline__ float rowmax16(float x) {
  x = fmaxf(x, __shfl_xor(x, 1, 32));
  x = fmaxf(x, __shfl_xor(x, 2, 32));
  x = fmaxf(x, __shfl_xor(x, 4, 32));
  x = fmaxf(x, __shfl_xor(x, 8, 32));
  return x;
}
__device__ __forceinline__ float rowsum16(float x) {
  x += __shfl_xor(x, 1, 32);
  x += __shfl_xor(x, 2, 32);
  x += __shfl_xor(x, 4, 32);
  x += __shfl_xor(x, 8, 32);
  return x;
}
#endif

// ---------------------------------------------------------------------------
// fp32 -> bf16 conversion, 8 elements/thread (f32x8 in, bf16x8 = b128 out)
// ---------------------------------------------------------------------------
__global__ void cvt_f32_bf16_v8(const float4* __restrict__ in,
                                uint4* __restrict__ out, int n8) {
  int i = blockIdx.x * blockDim.x + threadIdx.x;
  if (i < n8) {
    float4 a = in[2 * i], b = in[2 * i + 1];
    union { uint4 q; unsigned short u[8]; } o;
    o.u[0] = f2bf(a.x); o.u[1] = f2bf(a.y); o.u[2] = f2bf(a.z); o.u[3] = f2bf(a.w);
    o.u[4] = f2bf(b.x); o.u[5] = f2bf(b.y); o.u[6] = f2bf(b.z); o.u[7] = f2bf(b.w);
    out[i] = o.q;
  }
}

// ---------------------------------------------------------------------------
// C[M,N] = A[M,K] * W[N,K]^T   (A, W bf16 row-major; C bf16 or f32(+bias))
// Block tile 256x128, BK=32; 8 waves in 4(M)x2(N); wave tile 64x64 = 4x4 WMMA.
// Double-buffered LDS: one barrier per K-step; next tile's global loads are
// in flight while current tile computes.
// ---------------------------------------------------------------------------
template <bool OUT_BF16>
__global__ __launch_bounds__(256) void gemm_rt(
    const unsigned short* __restrict__ A,
    const unsigned short* __restrict__ W,
    void* __restrict__ Cout,
    const float* __restrict__ bias,
    int M, int N, int K) {
  constexpr int LDSS = 40;  // padded row stride (bf16 elems): conflict-free
  __shared__ __align__(16) unsigned short AsBuf[2][256 * LDSS];  // 2x20480 B
  __shared__ __align__(16) unsigned short BsBuf[2][128 * LDSS];  // 2x10240 B

  const int tid  = threadIdx.x;
  const int lane = tid & 31;
  const int w    = tid >> 5;
  const int r    = lane & 15;
  const int hf   = lane >> 4;
  const int wm   = w & 3;   // 0..3 -> 64-row slab
  const int wn   = w >> 2;  // 0..1 -> 64-col slab

  const int rowBase = blockIdx.y * 256;
  const int colBase = blockIdx.x * 128;

  v8f acc[4][4];
#pragma unroll
  for (int i = 0; i < 4; i++)
#pragma unroll
    for (int j = 0; j < 4; j++)
#pragma unroll
      for (int e = 0; e < 8; e++) acc[i][j][e] = 0.f;

  uint4 ra[4], rb[2];
  // ---- preload tile kb=0 and stage into buffer 0 ----
#pragma unroll
  for (int ld = 0; ld < 4; ld++) {
    int c = tid + ld * 256, row = c >> 2, c8 = (c & 3) * 8;
    ra[ld] = *(const uint4*)(A + (size_t)(rowBase + row) * K + c8);
  }
#pragma unroll
  for (int ld = 0; ld < 2; ld++) {
    int c = tid + ld * 256, row = c >> 2, c8 = (c & 3) * 8;
    rb[ld] = *(const uint4*)(W + (size_t)(colBase + row) * K + c8);
  }
#pragma unroll
  for (int ld = 0; ld < 4; ld++) {
    int c = tid + ld * 256, row = c >> 2, c8 = (c & 3) * 8;
    *(uint4*)&AsBuf[0][row * LDSS + c8] = ra[ld];
  }
#pragma unroll
  for (int ld = 0; ld < 2; ld++) {
    int c = tid + ld * 256, row = c >> 2, c8 = (c & 3) * 8;
    *(uint4*)&BsBuf[0][row * LDSS + c8] = rb[ld];
  }
  __syncthreads();

  int cur = 0;
  for (int kb = 0; kb < K; kb += 32) {
    const bool haveNext = (kb + 32) < K;
    if (haveNext) {
      // issue next tile's global loads (latency hidden behind WMMAs)
#pragma unroll
      for (int ld = 0; ld < 4; ld++) {
        int c = tid + ld * 256, row = c >> 2, c8 = (c & 3) * 8;
        ra[ld] = *(const uint4*)(A + (size_t)(rowBase + row) * K + kb + 32 + c8);
      }
#pragma unroll
      for (int ld = 0; ld < 2; ld++) {
        int c = tid + ld * 256, row = c >> 2, c8 = (c & 3) * 8;
        rb[ld] = *(const uint4*)(W + (size_t)(colBase + row) * K + kb + 32 + c8);
      }
      if (kb + 64 < K)  // pull tile-after-next toward L2
        __builtin_prefetch(A + (size_t)(rowBase + (tid >> 2)) * K + kb + 64, 0, 0);
    }

    const unsigned short* as = &AsBuf[cur][0];
    const unsigned short* bs = &BsBuf[cur][0];
    Frag af[4], bf[4];
#pragma unroll
    for (int i = 0; i < 4; i++) {
      int rr = (wm * 64 + i * 16 + r) * LDSS;
      af[i].q[0] = *(const uint4*)&as[rr + 8 * hf];
      af[i].q[1] = *(const uint4*)&as[rr + 16 + 8 * hf];
    }
#pragma unroll
    for (int j = 0; j < 4; j++) {
      int rr = (wn * 64 + j * 16 + r) * LDSS;
      bf[j].q[0] = *(const uint4*)&bs[rr + 8 * hf];
      bf[j].q[1] = *(const uint4*)&bs[rr + 16 + 8 * hf];
    }
#pragma unroll
    for (int i = 0; i < 4; i++)
#pragma unroll
      for (int j = 0; j < 4; j++) acc[i][j] = wmma_bf16(af[i], bf[j], acc[i][j]);

    if (haveNext) {
      // stage next tile into the other buffer (its readers synced last iter)
#pragma unroll
      for (int ld = 0; ld < 4; ld++) {
        int c = tid + ld * 256, row = c >> 2, c8 = (c & 3) * 8;
        *(uint4*)&AsBuf[cur ^ 1][row * LDSS + c8] = ra[ld];
      }
#pragma unroll
      for (int ld = 0; ld < 2; ld++) {
        int c = tid + ld * 256, row = c >> 2, c8 = (c & 3) * 8;
        *(uint4*)&BsBuf[cur ^ 1][row * LDSS + c8] = rb[ld];
      }
    }
    __syncthreads();
    cur ^= 1;
  }

  // Epilogue: C layout -> lane(r,hf), VGPR vi => (row vi+8*hf, col r)
#pragma unroll
  for (int i = 0; i < 4; i++) {
#pragma unroll
    for (int j = 0; j < 4; j++) {
      int col = colBase + wn * 64 + j * 16 + r;
      float bv = 0.f;
      if constexpr (!OUT_BF16) bv = bias ? bias[col] : 0.f;
#pragma unroll
      for (int vi = 0; vi < 8; vi++) {
        int row = rowBase + wm * 64 + i * 16 + vi + 8 * hf;
        float val = acc[i][j][vi] + bv;
        if constexpr (OUT_BF16)
          ((unsigned short*)Cout)[(size_t)row * N + col] = f2bf(val);
        else
          ((float*)Cout)[(size_t)row * N + col] = val;
      }
    }
  }
}

// ---------------------------------------------------------------------------
// Causal flash attention. Grid: (S/128, H, B), block 256 (8 waves).
// Each wave owns 16 query rows; Q fragments resident; K/V tiles (32 wide)
// double-buffered in LDS (one barrier per tile). PV contraction uses a
// permuted K-ordering (pos = 2*(kpos&15) | (kpos>>4)) applied identically to
// P and V stores (contractions are invariant under a shared K permutation),
// enabling packed b32 P writes.
// ---------------------------------------------------------------------------
__global__ __launch_bounds__(256) void flash_attn(
    const unsigned short* __restrict__ Qg,
    const unsigned short* __restrict__ Kg,
    const unsigned short* __restrict__ Vg,
    unsigned short* __restrict__ Og) {
  constexpr float KLOG = 0.18033688011112042f;  // log2(e) / sqrt(64)
  __shared__ __align__(16) unsigned short Ks[2][32 * 72];   // [kpos][d]
  __shared__ __align__(16) unsigned short Vts[2][64 * 40];  // [d][perm(kpos)]
  __shared__ __align__(16) unsigned short Ps[8][16 * 40];   // per-wave P tile

  const int tid  = threadIdx.x;
  const int lane = tid & 31;
  const int w    = tid >> 5;
  const int r    = lane & 15;
  const int hf   = lane >> 4;

  const int qb = blockIdx.x * 128;
  const int h  = blockIdx.y;
  const int b  = blockIdx.z;

  const int qw = qb + w * 16;  // this wave's first query row
  const size_t headBase = (size_t)b * SS * DD + (size_t)h * HD;

  // Resident Q fragments: A-layout, K dim = d (two 32-chunks of HD=64)
  Frag qf0, qf1;
  {
    const unsigned short* qp = Qg + headBase + (size_t)(qw + r) * DD;
    qf0.q[0] = *(const uint4*)(qp + 8 * hf);
    qf0.q[1] = *(const uint4*)(qp + 16 + 8 * hf);
    qf1.q[0] = *(const uint4*)(qp + 32 + 8 * hf);
    qf1.q[1] = *(const uint4*)(qp + 48 + 8 * hf);
  }

  float m[8], l[8];
  v8f O[4];
#pragma unroll
  for (int vi = 0; vi < 8; vi++) { m[vi] = -3.0e38f; l[vi] = 0.f; }
#pragma unroll
  for (int g = 0; g < 4; g++)
#pragma unroll
    for (int vi = 0; vi < 8; vi++) O[g][vi] = 0.f;

  const int krow  = tid >> 3;        // 0..31
  const int kcol8 = (tid & 7) * 8;   // 0..56
  const int vpos  = ((krow & 15) << 1) | (krow >> 4);  // permuted K position
  const int kEnd  = qb + 128;

  // ---- preload tile kc=0 into buffer 0 ----
  uint4 kreg, vreg;
  {
    const size_t gb = headBase + (size_t)krow * DD + kcol8;
    kreg = *(const uint4*)(Kg + gb);
    vreg = *(const uint4*)(Vg + gb);
    *(uint4*)&Ks[0][krow * 72 + kcol8] = kreg;
    union { uint4 q; unsigned short u[8]; } vv;
    vv.q = vreg;
#pragma unroll
    for (int e = 0; e < 8; e++) Vts[0][(kcol8 + e) * 40 + vpos] = vv.u[e];
  }
  __syncthreads();

  int cur = 0;
  for (int kc = 0; kc < kEnd; kc += 32) {
    const bool haveNext = (kc + 32) < kEnd;
    if (haveNext) {
      const size_t gb = headBase + (size_t)(kc + 32 + krow) * DD + kcol8;
      kreg = *(const uint4*)(Kg + gb);
      vreg = *(const uint4*)(Vg + gb);
      if (kc + 64 < kEnd) {  // pull tile-after-next toward L2
        __builtin_prefetch(Kg + headBase + (size_t)(kc + 64 + krow) * DD + kcol8, 0, 0);
        __builtin_prefetch(Vg + headBase + (size_t)(kc + 64 + krow) * DD + kcol8, 0, 0);
      }
    }

    if (kc < qw + 16) {  // skip once every later tile is fully masked
      const unsigned short* ks = &Ks[cur][0];
      const unsigned short* vs = &Vts[cur][0];
      // ---- scores: S = Q K^T (two 16x16 tiles, K-dim = HD in 2 steps) ----
      v8f s0, s1;
#pragma unroll
      for (int e = 0; e < 8; e++) { s0[e] = 0.f; s1[e] = 0.f; }
      {
        Frag kf;
        int rr = r * 72;
        kf.q[0] = *(const uint4*)&ks[rr + 8 * hf];
        kf.q[1] = *(const uint4*)&ks[rr + 16 + 8 * hf];
        s0 = wmma_bf16(qf0, kf, s0);
        kf.q[0] = *(const uint4*)&ks[rr + 32 + 8 * hf];
        kf.q[1] = *(const uint4*)&ks[rr + 48 + 8 * hf];
        s0 = wmma_bf16(qf1, kf, s0);
      }
      {
        Frag kf;
        int rr = (16 + r) * 72;
        kf.q[0] = *(const uint4*)&ks[rr + 8 * hf];
        kf.q[1] = *(const uint4*)&ks[rr + 16 + 8 * hf];
        s1 = wmma_bf16(qf0, kf, s1);
        kf.q[0] = *(const uint4*)&ks[rr + 32 + 8 * hf];
        kf.q[1] = *(const uint4*)&ks[rr + 48 + 8 * hf];
        s1 = wmma_bf16(qf1, kf, s1);
      }

      // ---- causal mask: only diagonal tiles need it (wave-uniform test) ----
      if (kc + 31 > qw) {
#pragma unroll
        for (int vi = 0; vi < 8; vi++) {
          int qg = qw + vi + 8 * hf;
          if (kc + r      > qg) s0[vi] = -3.0e38f;
          if (kc + 16 + r > qg) s1[vi] = -3.0e38f;
        }
      }

      // ---- online softmax (row = vi + 8*hf, reduced across 16 lanes) ------
      float c[8];
#pragma unroll
      for (int vi = 0; vi < 8; vi++) {
        float a0 = s0[vi], a1 = s1[vi];
        float mx = rowmax16(fmaxf(a0, a1));
        float mn = fmaxf(m[vi], mx);
        float cf = exp2f((m[vi] - mn) * KLOG);
        float p0 = exp2f((a0 - mn) * KLOG);
        float p1 = exp2f((a1 - mn) * KLOG);
        float ps = rowsum16(p0 + p1);
        l[vi] = l[vi] * cf + ps;
        m[vi] = mn;
        c[vi] = cf;
        // packed P store: (p0,p1) -> permuted positions 2r, 2r+1
        unsigned int pk = (unsigned)f2bf(p0) | ((unsigned)f2bf(p1) << 16);
        *(unsigned int*)&Ps[w][(vi + 8 * hf) * 40 + 2 * r] = pk;
      }
      asm volatile("s_wait_dscnt 0" ::: "memory");  // wave-local LDS RAW

      Frag pf;  // A-layout P fragment (16 q-rows x 32 permuted k-positions)
      pf.q[0] = *(const uint4*)&Ps[w][r * 40 + 8 * hf];
      pf.q[1] = *(const uint4*)&Ps[w][r * 40 + 16 + 8 * hf];

      // rescale running output by correction factor, then O += P * V
#pragma unroll
      for (int g = 0; g < 4; g++)
#pragma unroll
        for (int vi = 0; vi < 8; vi++) O[g][vi] *= c[vi];

#pragma unroll
      for (int g = 0; g < 4; g++) {
        Frag vf;
        int rr = (16 * g + r) * 40;
        vf.q[0] = *(const uint4*)&vs[rr + 8 * hf];
        vf.q[1] = *(const uint4*)&vs[rr + 16 + 8 * hf];
        O[g] = wmma_bf16(pf, vf, O[g]);
      }
    }

    if (haveNext) {  // stage next tile into the other buffer
      *(uint4*)&Ks[cur ^ 1][krow * 72 + kcol8] = kreg;
      union { uint4 q; unsigned short u[8]; } vv;
      vv.q = vreg;
#pragma unroll
      for (int e = 0; e < 8; e++)
        Vts[cur ^ 1][(kcol8 + e) * 40 + vpos] = vv.u[e];
    }
    __syncthreads();
    cur ^= 1;
  }

  // normalize + store ctx (bf16, [B,S,D] with heads packed in D)
#pragma unroll
  for (int vi = 0; vi < 8; vi++) {
    float inv = 1.0f / l[vi];
    size_t rowOff = headBase + (size_t)(qw + vi + 8 * hf) * DD;
#pragma unroll
    for (int g = 0; g < 4; g++)
      Og[rowOff + 16 * g + r] = f2bf(O[g][vi] * inv);
  }
}

// ---------------------------------------------------------------------------
extern "C" void kernel_launch(void* const* d_in, const int* in_sizes, int n_in,
                              void* d_out, int out_size, void* d_ws,
                              size_t ws_size, hipStream_t stream) {
  const float* x  = (const float*)d_in[0];
  const float* Wq = (const float*)d_in[1];
  const float* Wk = (const float*)d_in[2];
  const float* Wv = (const float*)d_in[3];
  const float* Wo = (const float*)d_in[4];
  const float* bo = (const float*)d_in[5];
  float* out = (float*)d_out;

  const int M = BB * SS;               // 8192 rows
  const size_t nx = (size_t)M * DD;    // 8,388,608
  const size_t nw = (size_t)DD * DD;   // 1,048,576

  unsigned short* xbf = (unsigned short*)d_ws;
  unsigned short* wqb = xbf + nx;
  unsigned short* wkb = wqb + nw;
  unsigned short* wvb = wkb + nw;
  unsigned short* wob = wvb + nw;
  unsigned short* qbf = wob + nw;
  unsigned short* kbf = qbf + nx;
  unsigned short* vbf = kbf + nx;
  unsigned short* cbf = vbf + nx;

  const int thr = 256;
  cvt_f32_bf16_v8<<<(int)(nx / 8 + thr - 1) / thr, thr, 0, stream>>>(
      (const float4*)x, (uint4*)xbf, (int)(nx / 8));
  cvt_f32_bf16_v8<<<(int)(nw / 8 + thr - 1) / thr, thr, 0, stream>>>(
      (const float4*)Wq, (uint4*)wqb, (int)(nw / 8));
  cvt_f32_bf16_v8<<<(int)(nw / 8 + thr - 1) / thr, thr, 0, stream>>>(
      (const float4*)Wk, (uint4*)wkb, (int)(nw / 8));
  cvt_f32_bf16_v8<<<(int)(nw / 8 + thr - 1) / thr, thr, 0, stream>>>(
      (const float4*)Wv, (uint4*)wvb, (int)(nw / 8));
  cvt_f32_bf16_v8<<<(int)(nw / 8 + thr - 1) / thr, thr, 0, stream>>>(
      (const float4*)Wo, (uint4*)wob, (int)(nw / 8));

  dim3 gg(DD / 128, M / 256);  // (8, 32)
  gemm_rt<true><<<gg, 256, 0, stream>>>(xbf, wqb, qbf, nullptr, M, DD, DD);
  gemm_rt<true><<<gg, 256, 0, stream>>>(xbf, wkb, kbf, nullptr, M, DD, DD);
  gemm_rt<true><<<gg, 256, 0, stream>>>(xbf, wvb, vbf, nullptr, M, DD, DD);

  dim3 ga(SS / 128, HH, BB);   // (16, 16, 4)
  flash_attn<<<ga, 256, 0, stream>>>(qbf, kbf, vbf, cbf);

  gemm_rt<false><<<gg, 256, 0, stream>>>(cbf, wob, out, bo, M, DD, DD);
}
